// HBlock_18597208391748
// MI455X (gfx1250) — compile-verified, matrix operand order checked by reading, not run on hardware
//
#include <hip/hip_runtime.h>
#include <hip/hip_bf16.h>

// ---------------------------------------------------------------------------
// Problem constants (from reference setup_inputs):
//   B=2, T=4160, D=1024, H=16, KVH=4, HD=64, update_start=2048, n_update=2112
// ---------------------------------------------------------------------------
#define Bq   2
#define Tq   4160
#define Dq   1024
#define Hq   16
#define KVHq 4
#define HDq  64
#define UPD  2048
#define NU   2112          // T - UPD
#define BT   (Bq*Tq)       // 8320 tokens total
#define BU   (Bq*NU)       // 4224 updated tokens

typedef __attribute__((ext_vector_type(16))) _Float16 v16h;
typedef __attribute__((ext_vector_type(8)))  _Float16 v8h;
typedef __attribute__((ext_vector_type(8)))  float    v8f;

#define WMMA(a,b,c) __builtin_amdgcn_wmma_f32_16x16x32_f16(false,(a),false,(b),(short)0,(c),false,false)

// Load a 16x32 A/B fragment for v_wmma_f32_16x16x32_f16 from a row-major
// operand whose contraction (K) dim is contiguous.  rowp = start of this
// lane's row (A: row m = lane&15; B: column n = lane&15), ko = K offset
// (multiple of 32), kg = (lane>>4)&1.  ISA layout: element e holds
// k = ko + (e>>3)*16 + kg*8 + (e&7)  -> two contiguous 16B chunks.
__device__ __forceinline__ v16h ld_frag(const _Float16* rowp, int ko, int kg) {
  v8h lo = *reinterpret_cast<const v8h*>(rowp + ko + kg * 8);
  v8h hi = *reinterpret_cast<const v8h*>(rowp + ko + 16 + kg * 8);
  v16h r;
#pragma unroll
  for (int i = 0; i < 8; ++i) { r[i] = lo[i]; r[i + 8] = hi[i]; }
  return r;
}

// ---------------------------------------------------------------------------
// 1) f32 -> f16 weight conversion
// ---------------------------------------------------------------------------
__global__ void k_cvt_f16(const float* __restrict__ s, _Float16* __restrict__ d, int n) {
  int i = blockIdx.x * blockDim.x + threadIdx.x;
  if (i < n) d[i] = (_Float16)s[i];
}

// ---------------------------------------------------------------------------
// 2) RMSNorm (per row of length D=1024) -> f16
// ---------------------------------------------------------------------------
__global__ __launch_bounds__(256) void k_rmsnorm(const float* __restrict__ x,
                                                 _Float16* __restrict__ y) {
  __shared__ float red[8];
  __shared__ float s_sc;
  int row = blockIdx.x;
  const float* xr = x + (size_t)row * Dq;
  float s = 0.f;
  for (int c = threadIdx.x; c < Dq; c += 256) { float v = xr[c]; s += v * v; }
  for (int m = 1; m < 32; m <<= 1) s += __shfl_xor(s, m, 32);
  int wave = threadIdx.x >> 5, lane = threadIdx.x & 31;
  if (lane == 0) red[wave] = s;
  __syncthreads();
  if (threadIdx.x == 0) {
    float t = 0.f;
    for (int i = 0; i < 8; ++i) t += red[i];
    s_sc = rsqrtf(t * (1.0f / Dq) + 1e-6f);
  }
  __syncthreads();
  float sc = s_sc;
  _Float16* yr = y + (size_t)row * Dq;
  for (int c = threadIdx.x; c < Dq; c += 256) yr[c] = (_Float16)(xr[c] * sc);
}

// ---------------------------------------------------------------------------
// 3) WMMA GEMM:  C[M,N] = A[M,K] (f16 row-major) * W[N,K]^T (f16 row-major)
//    Block = 256 thr = 8 waves (2x4) -> 128x256 block tile, 64x64 per wave.
//    Per k-step: 8 fragment loads (8 KB) feed 16 wmma -> 32 FLOP/B from L2.
//    mode 0: store f32.  mode 1: store f16 relu(x)^2 (FFN epilogue).
//    M % 128 == 0, N % 256 == 0, K % 32 == 0 for all call sites.
// ---------------------------------------------------------------------------
__global__ __launch_bounds__(256) void k_gemm(const _Float16* __restrict__ A,
                                              const _Float16* __restrict__ W,
                                              float* __restrict__ Cf,
                                              _Float16* __restrict__ Ch,
                                              int M, int N, int K, int mode) {
  int wave = threadIdx.x >> 5;
  int lane = threadIdx.x & 31;
  int lm = lane & 15, kg = (lane >> 4) & 1;
  int wm = wave >> 2;                 // 0..1
  int wn = wave & 3;                  // 0..3
  int row0 = blockIdx.y * 128 + wm * 64;
  int col0 = blockIdx.x * 256 + wn * 64;

  const _Float16* ap[4];
  const _Float16* bp[4];
#pragma unroll
  for (int i = 0; i < 4; ++i) {
    ap[i] = A + (size_t)(row0 + i * 16 + lm) * K;
    bp[i] = W + (size_t)(col0 + i * 16 + lm) * K;
  }

  v8f c[4][4];
#pragma unroll
  for (int i = 0; i < 4; ++i)
#pragma unroll
    for (int j = 0; j < 4; ++j)
      c[i][j] = (v8f){0.f,0.f,0.f,0.f,0.f,0.f,0.f,0.f};

  for (int ko = 0; ko < K; ko += 32) {
    v16h a[4], b[4];
#pragma unroll
    for (int i = 0; i < 4; ++i) a[i] = ld_frag(ap[i], ko, kg);
#pragma unroll
    for (int j = 0; j < 4; ++j) b[j] = ld_frag(bp[j], ko, kg);
#pragma unroll
    for (int i = 0; i < 4; ++i)
#pragma unroll
      for (int j = 0; j < 4; ++j)
        c[i][j] = WMMA(a[i], b[j], c[i][j]);
  }

  int hi = kg;                     // C layout: rows m = r + 8*hi, col n = lane&15
#pragma unroll
  for (int i = 0; i < 4; ++i) {
#pragma unroll
    for (int j = 0; j < 4; ++j) {
      int nn = col0 + j * 16 + lm;
#pragma unroll
      for (int r = 0; r < 8; ++r) {
        int mm = row0 + i * 16 + r + 8 * hi;
        float v = c[i][j][r];
        if (mode == 0) {
          Cf[(size_t)mm * N + nn] = v;
        } else {
          v = v > 0.f ? v * v : 0.f;
          Ch[(size_t)mm * N + nn] = (_Float16)v;
        }
      }
    }
  }
}

// ---------------------------------------------------------------------------
// 4) L2-normalize heads + relayout (f32 -> f16)
// ---------------------------------------------------------------------------
__global__ __launch_bounds__(32) void k_l2n_q(const float* __restrict__ qproj,
                                              _Float16* __restrict__ qh) {
  int idx = blockIdx.x;             // over B*H*NU
  int qi = idx % NU; int bh = idx / NU; int h = bh % Hq; int b = bh / Hq;
  int lane = threadIdx.x;
  const float* src = qproj + ((size_t)(b * Tq + UPD + qi)) * Dq + h * HDq;
  float v0 = src[lane], v1 = src[lane + 32];
  float s = v0 * v0 + v1 * v1;
  for (int m = 1; m < 32; m <<= 1) s += __shfl_xor(s, m, 32);
  float sc = 1.f / fmaxf(sqrtf(s), 1e-12f);
  _Float16* dst = qh + ((size_t)(b * Hq + h) * NU + qi) * HDq;
  dst[lane] = (_Float16)(v0 * sc);
  dst[lane + 32] = (_Float16)(v1 * sc);
}

__global__ __launch_bounds__(32) void k_l2n_k(const float* __restrict__ kproj,
                                              _Float16* __restrict__ kh) {
  int idx = blockIdx.x;             // over B*KVH*T
  int t = idx % Tq; int bk = idx / Tq; int kvh = bk % KVHq; int b = bk / KVHq;
  int lane = threadIdx.x;
  const float* src = kproj + ((size_t)(b * Tq + t)) * (KVHq * HDq) + kvh * HDq;
  float v0 = src[lane], v1 = src[lane + 32];
  float s = v0 * v0 + v1 * v1;
  for (int m = 1; m < 32; m <<= 1) s += __shfl_xor(s, m, 32);
  float sc = 1.f / fmaxf(sqrtf(s), 1e-12f);
  _Float16* dst = kh + ((size_t)(b * KVHq + kvh) * Tq + t) * HDq;
  dst[lane] = (_Float16)(v0 * sc);
  dst[lane + 32] = (_Float16)(v1 * sc);
}

// V: cast f32->f16 and store TRANSPOSED: vt[(b*KVH+kvh)*HD + d][T] so the
// P*V B-fragments read contiguous key runs.
__global__ __launch_bounds__(32) void k_v_t(const float* __restrict__ vproj,
                                            _Float16* __restrict__ vt) {
  int idx = blockIdx.x;             // over B*KVH*T
  int t = idx % Tq; int bk = idx / Tq; int kvh = bk % KVHq; int b = bk / KVHq;
  int lane = threadIdx.x;
  const float* src = vproj + ((size_t)(b * Tq + t)) * (KVHq * HDq) + kvh * HDq;
  _Float16* dst = vt + (size_t)(b * KVHq + kvh) * HDq * Tq;
  dst[(size_t)lane * Tq + t] = (_Float16)src[lane];
  dst[(size_t)(lane + 32) * Tq + t] = (_Float16)src[lane + 32];
}

// ---------------------------------------------------------------------------
// 5) Flash attention, GQA, causal (q positions UPD..T-1).
//    1 wave per 16-query tile per (b,h).  Key tiles of 32.
//    S: 4x wmma (16x16x32), P*V: 4x wmma into 16x64 f32 accumulators.
// ---------------------------------------------------------------------------
__global__ __launch_bounds__(32) void k_attn(const _Float16* __restrict__ qh,
                                             const _Float16* __restrict__ kh,
                                             const _Float16* __restrict__ vt,
                                             _Float16* __restrict__ attn) {
  __shared__ _Float16 Pl[16 * 32];      // P tile staging (row-major 16x32)
  const int NQT = NU / 16;              // 132 query tiles
  int qt = blockIdx.x % NQT;
  int bh = blockIdx.x / NQT;
  int h = bh % Hq, b = bh / Hq, kvh = h >> 2;   // rep = H/KVH = 4
  int lane = threadIdx.x, lm = lane & 15, hi = (lane >> 4) & 1, kg = hi;

  const _Float16* qrow = qh + ((size_t)(b * Hq + h) * NU + qt * 16 + lm) * HDq;
  v16h qa0 = ld_frag(qrow, 0, kg);
  v16h qa1 = ld_frag(qrow, 32, kg);

  v8f z = {0.f,0.f,0.f,0.f,0.f,0.f,0.f,0.f};
  v8f o0 = z, o1 = z, o2 = z, o3 = z;
  float mi[8], li[8];
#pragma unroll
  for (int r = 0; r < 8; ++r) { mi[r] = -INFINITY; li[r] = 0.f; }

  const _Float16* kbaseP = kh + (size_t)(b * KVHq + kvh) * Tq * HDq;
  const _Float16* vbaseP = vt + (size_t)(b * KVHq + kvh) * HDq * Tq;

  int qmax = UPD + qt * 16 + 15;
  int nkt = (qmax + 32) / 32;           // ceil((qmax+1)/32)

  for (int kt = 0; kt < nkt; ++kt) {
    int kv0 = kt * 32;
    // S = (Q K^T) * 1/sqrt(HD) for 16 queries x 32 keys
    v8f s0 = z, s1 = z;
    {
      const _Float16* k0 = kbaseP + (size_t)(kv0 + lm) * HDq;
      const _Float16* k1 = kbaseP + (size_t)(kv0 + 16 + lm) * HDq;
      v16h b00 = ld_frag(k0, 0, kg), b01 = ld_frag(k0, 32, kg);
      v16h b10 = ld_frag(k1, 0, kg), b11 = ld_frag(k1, 32, kg);
      s0 = WMMA(qa0, b00, s0);
      s0 = WMMA(qa1, b01, s0);
      s1 = WMMA(qa0, b10, s1);
      s1 = WMMA(qa1, b11, s1);
    }
    // online softmax; rows live across the 16 lanes of each half
    float al[8];
#pragma unroll
    for (int r = 0; r < 8; ++r) {
      int qpos = UPD + qt * 16 + r + 8 * hi;
      float x0 = s0[r] * 0.125f;
      float x1 = s1[r] * 0.125f;
      if (kv0 + lm > qpos)      x0 = -INFINITY;
      if (kv0 + 16 + lm > qpos) x1 = -INFINITY;
      float rm = fmaxf(x0, x1);
      for (int msk = 1; msk < 16; msk <<= 1) rm = fmaxf(rm, __shfl_xor(rm, msk, 32));
      float mn = fmaxf(mi[r], rm);
      float a = __expf(mi[r] - mn);               // exp(-inf)=0 on first tile
      float p0 = __expf(x0 - mn), p1 = __expf(x1 - mn);
      float rs = p0 + p1;
      for (int msk = 1; msk < 16; msk <<= 1) rs += __shfl_xor(rs, msk, 32);
      li[r] = li[r] * a + rs;
      mi[r] = mn;
      al[r] = a;
      int m = r + 8 * hi;
      Pl[m * 32 + lm] = (_Float16)p0;
      Pl[m * 32 + 16 + lm] = (_Float16)p1;
    }
#pragma unroll
    for (int r = 0; r < 8; ++r) { o0[r] *= al[r]; o1[r] *= al[r]; o2[r] *= al[r]; o3[r] *= al[r]; }
    __syncthreads();
    v16h pa = ld_frag(&Pl[lm * 32], 0, kg);       // re-layout P into an A fragment
    v16h vb0 = ld_frag(vbaseP + (size_t)(0  + lm) * Tq, kv0, kg);
    v16h vb1 = ld_frag(vbaseP + (size_t)(16 + lm) * Tq, kv0, kg);
    v16h vb2 = ld_frag(vbaseP + (size_t)(32 + lm) * Tq, kv0, kg);
    v16h vb3 = ld_frag(vbaseP + (size_t)(48 + lm) * Tq, kv0, kg);
    o0 = WMMA(pa, vb0, o0);
    o1 = WMMA(pa, vb1, o1);
    o2 = WMMA(pa, vb2, o2);
    o3 = WMMA(pa, vb3, o3);
    __syncthreads();
  }

  // write attn output (f16) in token-major layout for the Wo GEMM
#pragma unroll
  for (int r = 0; r < 8; ++r) {
    int m = r + 8 * hi;
    float inv = 1.f / li[r];
    size_t orow = ((size_t)b * NU + qt * 16 + m) * Dq + h * HDq;
    attn[orow + 0  + lm] = (_Float16)(o0[r] * inv);
    attn[orow + 16 + lm] = (_Float16)(o1[r] * inv);
    attn[orow + 32 + lm] = (_Float16)(o2[r] * inv);
    attn[orow + 48 + lm] = (_Float16)(o3[r] * inv);
  }
}

// ---------------------------------------------------------------------------
// 6) softcap tanh + residual:  x_new = x[:,UPD:] + 15*tanh(attn_out/15)
// ---------------------------------------------------------------------------
__global__ void k_softres(const float* __restrict__ x, const float* __restrict__ ao,
                          float* __restrict__ xn, int n) {
  int i = blockIdx.x * 256 + threadIdx.x;
  if (i >= n) return;
  int row = i >> 10, col = i & 1023;
  int b = row / NU, qi = row % NU;
  float t = 15.f * tanhf(ao[i] * (1.f / 15.f));
  xn[i] = x[((size_t)(b * Tq + UPD + qi) << 10) + col] + t;
}

// ---------------------------------------------------------------------------
// 7) final: out[:, :UPD] = x ; out[:, UPD:] = x_new + ffn
// ---------------------------------------------------------------------------
__global__ void k_final(const float* __restrict__ x, const float* __restrict__ xn,
                        const float* __restrict__ ffn, float* __restrict__ out, int n) {
  int i = blockIdx.x * 256 + threadIdx.x;
  if (i >= n) return;
  int col = i & 1023;
  int row = i >> 10;
  int b = row / Tq, t = row % Tq;
  if (t < UPD) {
    out[i] = x[i];
  } else {
    size_t j = ((size_t)b * NU + (t - UPD)) * Dq + col;
    out[i] = xn[j] + ffn[j];
  }
}

// ---------------------------------------------------------------------------
// Launcher
// ---------------------------------------------------------------------------
extern "C" void kernel_launch(void* const* d_in, const int* in_sizes, int n_in,
                              void* d_out, int out_size, void* d_ws, size_t ws_size,
                              hipStream_t stream) {
  const float* x     = (const float*)d_in[0];
  const float* Wq    = (const float*)d_in[1];
  const float* Wk    = (const float*)d_in[2];
  const float* Wv    = (const float*)d_in[3];
  const float* Wo    = (const float*)d_in[4];
  const float* Wfc   = (const float*)d_in[5];
  const float* Wproj = (const float*)d_in[6];
  float* out = (float*)d_out;

  char* ws = (char*)d_ws;
  size_t off = 0;
  auto alloc = [&](size_t bytes) { size_t o = off; off += (bytes + 255) & ~(size_t)255; return o; };

  _Float16* wq_h    = (_Float16*)(ws + alloc((size_t)Dq * Dq * 2));
  _Float16* wk_h    = (_Float16*)(ws + alloc((size_t)KVHq * HDq * Dq * 2));
  _Float16* wv_h    = (_Float16*)(ws + alloc((size_t)KVHq * HDq * Dq * 2));
  _Float16* wo_h    = (_Float16*)(ws + alloc((size_t)Dq * Dq * 2));
  _Float16* wfc_h   = (_Float16*)(ws + alloc((size_t)4 * Dq * Dq * 2));
  _Float16* wproj_h = (_Float16*)(ws + alloc((size_t)4 * Dq * Dq * 2));
  _Float16* xnorm_h = (_Float16*)(ws + alloc((size_t)BT * Dq * 2));
  float*    qproj_f = (float*)   (ws + alloc((size_t)BT * Dq * 4));
  float*    kproj_f = (float*)   (ws + alloc((size_t)BT * KVHq * HDq * 4));
  float*    vproj_f = (float*)   (ws + alloc((size_t)BT * KVHq * HDq * 4));
  _Float16* q_h     = (_Float16*)(ws + alloc((size_t)BU * Dq * 2));
  _Float16* k_h     = (_Float16*)(ws + alloc((size_t)BT * KVHq * HDq * 2));
  _Float16* vt_h    = (_Float16*)(ws + alloc((size_t)BT * KVHq * HDq * 2));
  _Float16* attn_h  = (_Float16*)(ws + alloc((size_t)BU * Dq * 2));
  float*    attno_f = (float*)   (ws + alloc((size_t)BU * Dq * 4));
  float*    xnew_f  = (float*)   (ws + alloc((size_t)BU * Dq * 4));
  _Float16* xn2_h   = (_Float16*)(ws + alloc((size_t)BU * Dq * 2));
  _Float16* h_h     = (_Float16*)(ws + alloc((size_t)BU * 4 * Dq * 2));
  float*    ffn_f   = (float*)   (ws + alloc((size_t)BU * Dq * 4));
  (void)ws_size; (void)n_in; (void)in_sizes; (void)out_size;

  auto cvt = [&](const float* s, _Float16* d, int n) {
    k_cvt_f16<<<(n + 255) / 256, 256, 0, stream>>>(s, d, n);
  };
  cvt(Wq, wq_h, Dq * Dq);
  cvt(Wk, wk_h, KVHq * HDq * Dq);
  cvt(Wv, wv_h, KVHq * HDq * Dq);
  cvt(Wo, wo_h, Dq * Dq);
  cvt(Wfc, wfc_h, 4 * Dq * Dq);
  cvt(Wproj, wproj_h, 4 * Dq * Dq);

  // RMSNorm all tokens -> f16
  k_rmsnorm<<<BT, 256, 0, stream>>>(x, xnorm_h);

  // Projections (q computed for all tokens; only updated rows consumed)
  k_gemm<<<dim3(Dq / 256, BT / 128), 256, 0, stream>>>(xnorm_h, wq_h, qproj_f, nullptr, BT, Dq, Dq, 0);
  k_gemm<<<dim3((KVHq * HDq) / 256, BT / 128), 256, 0, stream>>>(xnorm_h, wk_h, kproj_f, nullptr, BT, KVHq * HDq, Dq, 0);
  k_gemm<<<dim3((KVHq * HDq) / 256, BT / 128), 256, 0, stream>>>(xnorm_h, wv_h, vproj_f, nullptr, BT, KVHq * HDq, Dq, 0);

  k_l2n_q<<<Bq * Hq * NU, 32, 0, stream>>>(qproj_f, q_h);
  k_l2n_k<<<Bq * KVHq * Tq, 32, 0, stream>>>(kproj_f, k_h);
  k_v_t  <<<Bq * KVHq * Tq, 32, 0, stream>>>(vproj_f, vt_h);

  // Attention
  k_attn<<<Bq * Hq * (NU / 16), 32, 0, stream>>>(q_h, k_h, vt_h, attn_h);

  // Output projection + softcap + residual
  k_gemm<<<dim3(Dq / 256, BU / 128), 256, 0, stream>>>(attn_h, wo_h, attno_f, nullptr, BU, Dq, Dq, 0);
  k_softres<<<(BU * Dq + 255) / 256, 256, 0, stream>>>(x, attno_f, xnew_f, BU * Dq);

  // FFN
  k_rmsnorm<<<BU, 256, 0, stream>>>(xnew_f, xn2_h);
  k_gemm<<<dim3((4 * Dq) / 256, BU / 128), 256, 0, stream>>>(xn2_h, wfc_h, nullptr, h_h, BU, 4 * Dq, Dq, 1);
  k_gemm<<<dim3(Dq / 256, BU / 128), 256, 0, stream>>>(h_h, wproj_h, ffn_f, nullptr, BU, Dq, 4 * Dq, 0);

  // Final assembly (prefix copy + second residual)
  k_final<<<(Bq * Tq * Dq + 255) / 256, 256, 0, stream>>>(x, xnew_f, ffn_f, out, Bq * Tq * Dq);
}
